// DualTowerTitans_18743237279814
// MI455X (gfx1250) — compile-verified, matrix operand order checked by reading, not run on hardware
//
#include <hip/hip_runtime.h>
#include <hip/hip_bf16.h>
#include <stdint.h>

// ---------------------------------------------------------------------------
// DualTowerTitans on MI455X (gfx1250, wave32, WMMA + TDM)
//   B=512 T=64 D=512 H=8 DH=64 FF=2048
//   - 32 workgroups x 512 threads (16 wave32 = 4/SIMD); each owns a 16-row
//     batch tile and its slice of the (B,H,64,64) fp32 state (L2-resident).
//   - GEMMs: v_wmma_f32_16x16x32_f16, A staged in LDS, B pre-swizzled f16
//     fragments (one contiguous 32B load per lane).
//   - Item tiles prefetched one step ahead via TENSOR_LOAD_TO_LDS (TDM,
//     TENSORcnt); one-shot tiles via global_load_async_to_lds_b128 (ASYNCcnt).
// ---------------------------------------------------------------------------

#define DEV __device__ __forceinline__

typedef _Float16 v16h __attribute__((ext_vector_type(16)));
typedef float    v8f  __attribute__((ext_vector_type(8)));
typedef unsigned int v4u __attribute__((ext_vector_type(4)));
typedef int          v8i __attribute__((ext_vector_type(8)));
typedef int          v4i __attribute__((ext_vector_type(4)));

static constexpr int NTH = 512;       // threads per block (16 wave32)
static constexpr int NWV = 16;        // waves per block

// ---------------- CDNA5 async global->LDS copy (ASYNCcnt) ------------------
DEV void async_ld_b128(void* lds_dst, const void* gsrc) {
    uint32_t ldsoff = (uint32_t)(uintptr_t)lds_dst;
    unsigned long long ga = (unsigned long long)(uintptr_t)gsrc;
    asm volatile("global_load_async_to_lds_b128 %0, %1, off"
                 :: "v"(ldsoff), "v"(ga) : "memory");
}
DEV void async_wait_all() {
    asm volatile("s_wait_asynccnt 0x0" ::: "memory");
}

// ---------------- TDM: 2D tile (16 rows x 512 f32) -> LDS (TENSORcnt) ------
// D# per CDNA5 ISA: group0 {count=1, lds_addr,
// global_addr, type=2}; group1 {data_size=4B, tensor_dim0=512,
// tensor_dim1=32768, tile_dim0=512, tile_dim1=16, dim0_stride=T*D=32768};
// remaining groups zero (<=2D tensor). Toolchain uses the 6-arg builtin.
DEV void tdm_load_tile16x512(void* lds_dst, const void* gsrc) {
    unsigned long long ga = (unsigned long long)(uintptr_t)gsrc;
    v4u g0;
    g0.x = 1u;                                    // count=1 (valid descriptor)
    g0.y = (uint32_t)(uintptr_t)lds_dst;          // lds_addr
    g0.z = (uint32_t)ga;                          // global_addr[31:0]
    g0.w = (uint32_t)(ga >> 32) | (2u << 30);     // global_addr[56:32] | type=2
    v8i g1;
    g1[0] = 0x00020000;                           // wg_mask=0, data_size=2 (4B)
    g1[1] = (512 << 16);                          // tensor_dim0 = 512
    g1[2] = (int)0x80000000u;                     // tensor_dim1 = 32768 (B*T)
    g1[3] = (512 << 16);                          // tile_dim0 = 512
    g1[4] = 16;                                   // tile_dim1 = 16
    g1[5] = 32768;                                // tensor_dim0_stride = T*D
    g1[6] = 0; g1[7] = 0;
    v4i z4 = {0, 0, 0, 0};
    v8i z8 = {0, 0, 0, 0, 0, 0, 0, 0};
    __builtin_amdgcn_tensor_load_to_lds(g0, g1, z4, z4, z8, 0);
}
DEV void tdm_wait_all() { __builtin_amdgcn_s_wait_tensorcnt(0); }

// ---------------- WMMA tile (M=16, N=16, K-loop of 32) ---------------------
DEV v8f wmma_tile(const _Float16* Alds, int ldK, const _Float16* Bfrag,
                  int ktiles, int ntile, int nTilesTotal, int lane) {
    v8f acc = {0.f,0.f,0.f,0.f,0.f,0.f,0.f,0.f};
    const int m  = lane & 15;
    const int kb = (lane >> 4) << 3;
    for (int kt = 0; kt < ktiles; ++kt) {
        union { v16h v; uint32_t u[8]; } a;
        const uint32_t* arow = (const uint32_t*)(Alds + m * ldK + kt * 32);
#pragma unroll
        for (int p = 0; p < 8; ++p) {                // ISA 16-bit A layout
            int kk = ((p >> 2) << 4) + kb + ((p & 3) << 1);
            a.u[p] = arow[kk >> 1];
        }
        const v16h* bp = (const v16h*)Bfrag + (size_t)(kt * nTilesTotal + ntile) * 32 + lane;
        v16h b = *bp;
        acc = __builtin_amdgcn_wmma_f32_16x16x32_f16(
                  false, a.v, false, b, (short)0, acc, false, false);
    }
    return acc;
}

DEV void store_tile_f32(float* Out, int ldN, int n0, v8f acc,
                        const float* bias, int lane) {
    int n  = n0 + (lane & 15);
    int mb = (lane >> 4) << 3;
    float bv = bias ? bias[n] : 0.f;
#pragma unroll
    for (int r = 0; r < 8; ++r) Out[(mb + r) * ldN + n] = acc[r] + bv;
}
DEV void store_tile_f16(_Float16* Out, int ldN, int n0, v8f acc,
                        const float* bias, bool relu, int lane) {
    int n  = n0 + (lane & 15);
    int mb = (lane >> 4) << 3;
    float bv = bias ? bias[n] : 0.f;
#pragma unroll
    for (int r = 0; r < 8; ++r) {
        float v = acc[r] + bv;
        if (relu) v = fmaxf(v, 0.f);
        Out[(mb + r) * ldN + n] = (_Float16)v;
    }
}

DEV void gemm_f32out(const _Float16* A, int K, const _Float16* Bfrag, int N,
                     float* Out, const float* bias, int tid) {
    int wave = tid >> 5, lane = tid & 31;
    int nT = N >> 4, kT = K >> 5;
    for (int nt = wave; nt < nT; nt += NWV) {
        v8f acc = wmma_tile(A, K, Bfrag, kT, nt, nT, lane);
        store_tile_f32(Out, N, nt << 4, acc, bias, lane);
    }
}
DEV void gemm_f16out(const _Float16* A, int K, const _Float16* Bfrag, int N,
                     _Float16* Out, int outLd, int outOff,
                     const float* bias, bool relu, int tid) {
    int wave = tid >> 5, lane = tid & 31;
    int nT = N >> 4, kT = K >> 5;
    for (int nt = wave; nt < nT; nt += NWV) {
        v8f acc = wmma_tile(A, K, Bfrag, kT, nt, nT, lane);
        store_tile_f16(Out, outLd, outOff + (nt << 4), acc, bias, relu, lane);
    }
}

// ---------------- LayerNorm over 16 rows of 512 (one wave per row) ---------
DEV void ln16x512(const float* Xin, float* out32, _Float16* out16, int ld16,
                  const float* gw, const float* bw, int tid, bool relu) {
    int wave = tid >> 5, lane = tid & 31;
    int r = wave;                                // 16 waves, 16 rows
    const float* x = Xin + r * 512;
    float s = 0.f, s2 = 0.f;
    for (int c = lane; c < 512; c += 32) { float v = x[c]; s += v; s2 = fmaf(v, v, s2); }
#pragma unroll
    for (int off = 16; off > 0; off >>= 1) {
        s  += __shfl_xor(s,  off, 32);
        s2 += __shfl_xor(s2, off, 32);
    }
    float mean = s * (1.f / 512.f);
    float var  = s2 * (1.f / 512.f) - mean * mean;
    float rstd = rsqrtf(var + 1e-5f);
    for (int c = lane; c < 512; c += 32) {
        float v = (x[c] - mean) * rstd * gw[c] + bw[c];
        if (relu)  v = fmaxf(v, 0.f);
        if (out32) out32[r * 512 + c] = v;
        if (out16) out16[r * ld16 + c] = (_Float16)v;
    }
}

DEV void l2n_heads(float* X, int tid) {
    if (tid < 128) {
        int r = tid >> 3, h = tid & 7;
        float* x = X + r * 512 + h * 64;
        float ss = 0.f;
#pragma unroll 8
        for (int j = 0; j < 64; ++j) ss = fmaf(x[j], x[j], ss);
        float inv = 1.f / fmaxf(sqrtf(ss), 1e-12f);
#pragma unroll 8
        for (int j = 0; j < 64; ++j) x[j] *= inv;
    }
}

// out[b,h,i] = sum_j state[b,h,i,j] * vec[b,h,j]
DEV void state_matvec(const float* state, const float* Vec,
                      float* out32, _Float16* out16, int o16ld, int o16off,
                      int g, int tid) {
    for (int s = 0; s < 16; ++s) {
        int task = tid + s * NTH;            // 16*512 tasks
        int r = task >> 9, col = task & 511;
        int h = col >> 6, i = col & 63;
        const float4* srow = (const float4*)(state +
            ((((size_t)(g * 16 + r)) * 8 + h) * 64 + i) * 64);
        const float* q = Vec + r * 512 + h * 64;
        float acc = 0.f;
#pragma unroll
        for (int j4 = 0; j4 < 16; ++j4) {
            float4 sv = srow[j4];
            acc = fmaf(sv.x, q[j4 * 4 + 0], acc);
            acc = fmaf(sv.y, q[j4 * 4 + 1], acc);
            acc = fmaf(sv.z, q[j4 * 4 + 2], acc);
            acc = fmaf(sv.w, q[j4 * 4 + 3], acc);
        }
        if (out32) out32[r * 512 + col] = acc;
        if (out16) out16[r * o16ld + o16off + col] = (_Float16)acc;
    }
}

// state = (1-alpha)*state + eta*(v-pred) (x) k
DEV void state_update(float* state, const float* Klds, const _Float16* V16,
                      const float* PRED, const float* AL, const float* ET,
                      int g, int tid) {
    for (int s = 0; s < 16; ++s) {
        int task = tid + s * NTH;
        int r = task >> 9, col = task & 511;
        int h = col >> 6, i = col & 63;
        float a  = AL[r * 8 + h];
        float e  = ET[r * 8 + h];
        float d  = e * ((float)V16[r * 1024 + 512 + col] - PRED[r * 512 + col]);
        float om = 1.f - a;
        float4* srow = (float4*)(state +
            ((((size_t)(g * 16 + r)) * 8 + h) * 64 + i) * 64);
        const float* k = Klds + r * 512 + h * 64;
#pragma unroll
        for (int j4 = 0; j4 < 16; ++j4) {
            float4 sv = srow[j4];
            sv.x = om * sv.x + d * k[j4 * 4 + 0];
            sv.y = om * sv.y + d * k[j4 * 4 + 1];
            sv.z = om * sv.z + d * k[j4 * 4 + 2];
            sv.w = om * sv.w + d * k[j4 * 4 + 3];
            srow[j4] = sv;
        }
    }
}

// alpha/eta gate matvecs; 256 tasks
DEV void alpha_eta(const float* MOD, const _Float16* MO16,
                   const float* Wa, const float* ba,
                   const float* We, const float* be,
                   float* AL, float* ET, int tid) {
    if (tid < 256) {
        int which = tid >> 7;                   // 0: alpha, 1: eta
        int r = (tid >> 3) & 15, h = tid & 7;
        const float* W  = which ? We : Wa;
        const float* bb = which ? be : ba;
        const float* m  = MOD + r * 512;
        float z = 0.f;
        for (int c = 0; c < 512; ++c) z = fmaf(m[c], W[c * 8 + h], z);
        z += bb[h] + (float)MO16[r * 1040 + 1024 + which * 8 + h];
        float sg = 1.f / (1.f + __expf(-z));
        if (which) ET[r * 8 + h] = sg * 0.125f; // SCALE = DH^-0.5
        else       AL[r * 8 + h] = sg;
    }
}

// ---------------- weight -> f16 WMMA-B fragment swizzle --------------------
__global__ void frag_kernel(const float* __restrict__ W, _Float16* __restrict__ out,
                            int K, int N) {
    int total = K * N;
    int nT = N >> 4;
    for (int idx = blockIdx.x * blockDim.x + threadIdx.x; idx < total;
         idx += gridDim.x * blockDim.x) {
        int j    = idx & 15;
        int lane = (idx >> 4) & 31;
        int tile = idx >> 9;
        int nt = tile % nT, kt = tile / nT;
        int p = j >> 1, half = j & 1;
        int k = kt * 32 + ((p >> 2) << 4) + ((lane >> 4) << 3) + ((p & 3) << 1) + half;
        int n = nt * 16 + (lane & 15);
        out[idx] = (_Float16)W[(size_t)k * N + n];
    }
}

// ---------------- main recurrent scan --------------------------------------
__global__ __launch_bounds__(NTH) void scan_kernel(
    const float* __restrict__ item_seq, const float* __restrict__ user_static,
    float* __restrict__ state,
    const _Float16* __restrict__ fWq, const _Float16* __restrict__ fWk,
    const _Float16* __restrict__ fWv, const _Float16* __restrict__ fmcW1,
    const _Float16* __restrict__ fmcW2,
    const float* __restrict__ Wa, const float* __restrict__ ba,
    const float* __restrict__ We, const float* __restrict__ be,
    const float* __restrict__ mcb1, const float* __restrict__ mcg,
    const float* __restrict__ mcbt, const float* __restrict__ mcb2,
    const float* __restrict__ n1g, const float* __restrict__ n1b) {

    __shared__ float    PF   [16 * 512];    // TDM prefetch buffer (next item tile)
    __shared__ _Float16 A512 [16 * 512];    // GEMM A staging (K=512)
    __shared__ _Float16 A1024[16 * 1024];   // meta_in = [us_norm | ctx]; v reuses cols 512+
    __shared__ float    INORM[16 * 512];    // i_norm -> mod -> pred
    __shared__ _Float16 MO16 [16 * 1040];   // meta output (gamma|beta|biases)
    __shared__ float    X32  [16 * 512];    // q_raw/q, h_raw, k_raw/k
    __shared__ float    AL[128], ET[128];

    const int tid = threadIdx.x, g = blockIdx.x;

    // us_norm -> A1024[:, :512] (resident across steps); async-to-LDS path
    for (int it = 0; it < 4; ++it) {
        int idx = tid + it * NTH;            // b128 index over 16x512 f32
        int r = idx >> 7, c4 = idx & 127;
        async_ld_b128(&INORM[r * 512 + c4 * 4],
                      user_static + (size_t)(g * 16 + r) * 512 + c4 * 4);
    }
    async_wait_all();
    __syncthreads();
    ln16x512(INORM, nullptr, A1024, 1024, n1g, n1b, tid, false);

    // TDM prefetch of item tile t=0 (wave 0 only; descriptor-driven DMA)
    if (tid < 32) {
        tdm_load_tile16x512(PF, item_seq + ((size_t)(g * 16) * 64 + 0) * 512);
    }
    __syncthreads();

    for (int t = 0; t < 64; ++t) {
        // P0: consume prefetched tile
        tdm_wait_all();
        __syncthreads();
        // P1: i_norm = LN(item)  (PF -> INORM f32, A512 f16)
        ln16x512(PF, INORM, A512, 512, n1g, n1b, tid, false);
        __syncthreads();
        // prefetch next tile while this step computes
        if (t + 1 < 64 && tid < 32) {
            tdm_load_tile16x512(PF, item_seq + ((size_t)(g * 16) * 64 + (t + 1)) * 512);
        }
        // P2: q_raw = i_norm @ Wq
        gemm_f32out(A512, 512, fWq, 512, X32, nullptr, tid);
        __syncthreads();
        // P3: q = l2n(q_raw) per head
        l2n_heads(X32, tid);
        __syncthreads();
        // P4: mem_ctx = state @ q  -> A1024[:, 512:]
        state_matvec(state, X32, nullptr, A1024, 1024, 512, g, tid);
        __syncthreads();
        // P5: h = relu(LN(meta_in @ mcW1 + mcb1))
        gemm_f32out(A1024, 1024, fmcW1, 512, X32, mcb1, tid);
        __syncthreads();
        ln16x512(X32, nullptr, A512, 512, mcg, mcbt, tid, true);
        __syncthreads();
        // P6: mo = h @ mcW2 + mcb2 (16x1040, f16)
        gemm_f16out(A512, 512, fmcW2, 1040, MO16, 1040, 0, mcb2, false, tid);
        __syncthreads();
        // P7: mod = i_norm*(1+tanh(gamma)) + beta
        for (int s = 0; s < 16; ++s) {
            int idx = tid + s * NTH;
            int r = idx >> 9, c = idx & 511;
            float gam = (float)MO16[r * 1040 + c];
            float bet = (float)MO16[r * 1040 + 512 + c];
            float mv  = INORM[r * 512 + c] * (1.f + tanhf(gam)) + bet;
            INORM[r * 512 + c] = mv;
            A512[r * 512 + c]  = (_Float16)mv;
        }
        __syncthreads();
        // P8: alpha/eta gates
        alpha_eta(INORM, MO16, Wa, ba, We, be, AL, ET, tid);
        __syncthreads();
        // P9: k_raw = mod @ Wk ; v = mod @ Wv (f16 into A1024[:,512:])
        gemm_f32out(A512, 512, fWk, 512, X32, nullptr, tid);
        gemm_f16out(A512, 512, fWv, 512, A1024, 1024, 512, nullptr, false, tid);
        __syncthreads();
        // P10: k = l2n(k_raw)
        l2n_heads(X32, tid);
        __syncthreads();
        // P11: pred = state @ k -> INORM (reuse)
        state_matvec(state, X32, INORM, nullptr, 0, 0, g, tid);
        __syncthreads();
        // P12: state = (1-alpha)*state + eta*(v-pred)(x)k
        state_update(state, X32, A1024, INORM, AL, ET, g, tid);
        __syncthreads();
    }
}

// ---------------- final read + output projection + FFN ---------------------
__global__ __launch_bounds__(NTH) void final_kernel(
    const float* __restrict__ last_item, const float* __restrict__ user_static,
    const float* __restrict__ state,
    const _Float16* __restrict__ fWq, const _Float16* __restrict__ fWo,
    const _Float16* __restrict__ fW1, const _Float16* __restrict__ fW2,
    const float* __restrict__ bo, const float* __restrict__ b1,
    const float* __restrict__ b2,
    const float* __restrict__ n1g, const float* __restrict__ n1b,
    const float* __restrict__ n2g, const float* __restrict__ n2b,
    float* __restrict__ out) {

    __shared__ _Float16 A512[16 * 512];
    __shared__ _Float16 T16 [16 * 2048];     // relu(x2@W1) f16
    __shared__ float    L32 [16 * 512];      // x (kept for residual)
    __shared__ float    X32 [16 * 512];

    const int tid = threadIdx.x, g = blockIdx.x;

    for (int it = 0; it < 4; ++it) {
        int idx = tid + it * NTH;
        int r = idx >> 7, c4 = idx & 127;
        async_ld_b128(&L32[r * 512 + c4 * 4],
                      last_item + (size_t)(g * 16 + r) * 512 + c4 * 4);
    }
    async_wait_all();
    __syncthreads();
    ln16x512(L32, nullptr, A512, 512, n1g, n1b, tid, false);   // q_norm
    __syncthreads();
    gemm_f32out(A512, 512, fWq, 512, X32, nullptr, tid);
    __syncthreads();
    l2n_heads(X32, tid);
    __syncthreads();
    state_matvec(state, X32, nullptr, A512, 512, 0, g, tid);   // read -> f16
    __syncthreads();
    gemm_f32out(A512, 512, fWo, 512, X32, bo, tid);            // attn
    __syncthreads();
    for (int s = 0; s < 16; ++s) {                             // x = user + attn
        int idx = tid + s * NTH;
        int r = idx >> 9, c = idx & 511;
        L32[r * 512 + c] = user_static[(size_t)(g * 16 + r) * 512 + c]
                         + X32[r * 512 + c];
    }
    __syncthreads();
    ln16x512(L32, nullptr, A512, 512, n2g, n2b, tid, false);   // x2
    __syncthreads();
    gemm_f16out(A512, 512, fW1, 2048, T16, 2048, 0, b1, true, tid);
    __syncthreads();
    gemm_f32out(T16, 2048, fW2, 512, X32, b2, tid);            // ffn
    __syncthreads();
    for (int s = 0; s < 16; ++s) {
        int idx = tid + s * NTH;
        int r = idx >> 9, c = idx & 511;
        out[(size_t)(g * 16 + r) * 512 + c] = L32[r * 512 + c] + X32[r * 512 + c];
    }
}

// ---------------------------------------------------------------------------
extern "C" void kernel_launch(void* const* d_in, const int* in_sizes, int n_in,
                              void* d_out, int out_size, void* d_ws, size_t ws_size,
                              hipStream_t stream) {
    (void)in_sizes; (void)n_in; (void)out_size; (void)ws_size;
    const float* item_seq   = (const float*)d_in[0];
    const float* user_stat  = (const float*)d_in[1];
    const float* last_item  = (const float*)d_in[2];
    const float* init_state = (const float*)d_in[3];
    const float* Wq  = (const float*)d_in[4];
    const float* Wk  = (const float*)d_in[5];
    const float* Wv  = (const float*)d_in[6];
    const float* Wa  = (const float*)d_in[7];
    const float* ba  = (const float*)d_in[8];
    const float* We  = (const float*)d_in[9];
    const float* be  = (const float*)d_in[10];
    const float* mcW1= (const float*)d_in[11];
    const float* mcb1= (const float*)d_in[12];
    const float* mcg = (const float*)d_in[13];
    const float* mcbt= (const float*)d_in[14];
    const float* mcW2= (const float*)d_in[15];
    const float* mcb2= (const float*)d_in[16];
    const float* Wo  = (const float*)d_in[17];
    const float* bo  = (const float*)d_in[18];
    const float* W1  = (const float*)d_in[19];
    const float* b1  = (const float*)d_in[20];
    const float* W2  = (const float*)d_in[21];
    const float* b2  = (const float*)d_in[22];
    const float* n1g = (const float*)d_in[23];
    const float* n1b = (const float*)d_in[24];
    const float* n2g = (const float*)d_in[25];
    const float* n2b = (const float*)d_in[26];
    float* out = (float*)d_out;

    // workspace: [ state f32 (67MB) | f16 fragment weights (~8.4MB) ]
    float* state = (float*)d_ws;
    size_t stateElems = (size_t)512 * 8 * 64 * 64;
    _Float16* fp = (_Float16*)((char*)d_ws + stateElems * sizeof(float));
    _Float16* fWq   = fp; fp += 512 * 512;
    _Float16* fWk   = fp; fp += 512 * 512;
    _Float16* fWv   = fp; fp += 512 * 512;
    _Float16* fmcW1 = fp; fp += 1024 * 512;
    _Float16* fmcW2 = fp; fp += 512 * 1040;
    _Float16* fWo   = fp; fp += 512 * 512;
    _Float16* fW1   = fp; fp += 512 * 2048;
    _Float16* fW2   = fp; fp += 2048 * 512;

    (void)hipMemcpyAsync(state, init_state, stateElems * sizeof(float),
                         hipMemcpyDeviceToDevice, stream);

    struct FJ { const float* W; _Float16* f; int K, N; };
    FJ jobs[8] = {
        {Wq, fWq, 512, 512},   {Wk, fWk, 512, 512},   {Wv, fWv, 512, 512},
        {mcW1, fmcW1, 1024, 512}, {mcW2, fmcW2, 512, 1040},
        {Wo, fWo, 512, 512},   {W1, fW1, 512, 2048},  {W2, fW2, 2048, 512},
    };
    for (int i = 0; i < 8; ++i) {
        int total = jobs[i].K * jobs[i].N;
        int blocks = (total + 255) / 256;
        if (blocks > 2048) blocks = 2048;
        frag_kernel<<<blocks, 256, 0, stream>>>(jobs[i].W, jobs[i].f,
                                                jobs[i].K, jobs[i].N);
    }

    scan_kernel<<<32, NTH, 0, stream>>>(item_seq, user_stat, state,
        fWq, fWk, fWv, fmcW1, fmcW2,
        Wa, ba, We, be, mcb1, mcg, mcbt, mcb2, n1g, n1b);

    final_kernel<<<32, NTH, 0, stream>>>(last_item, user_stat, state,
        fWq, fWo, fW1, fW2, bo, b1, b2, n1g, n1b, n2g, n2b, out);
}
